// SwinAttention_11622181503353
// MI455X (gfx1250) — matrix operand order repaired and checked by reading, not compiled, not run on hardware
//
#include <hip/hip_runtime.h>
#include <hip/hip_bf16.h>

typedef _Float16 v16h __attribute__((ext_vector_type(16)));
typedef _Float16 v8h  __attribute__((ext_vector_type(8)));
typedef _Float16 v4h  __attribute__((ext_vector_type(4)));
typedef float    v8f  __attribute__((ext_vector_type(8)));

#define DIMD     256
#define TSEQ     8192
#define WS       8
#define SHIFTV   4
#define HEADS    8
#define HDIM     32
#define NWIN_TOT 8192   // B * (T / WS) = 8 * 1024
#define WPB      4      // windows per workgroup
#define ROWS     32     // WPB * WS
#define KTILES   8      // 256 / 32
#define NT_QKV   48     // 768 / 16
#define NT_PROJ  16     // 256 / 16

// ---------------------------------------------------------------------------
// Pre-pass: pack f32 weight matrix [K, ldn] into f16 WMMA B-fragments.
// Tile (kt, nt) is 512 halves: lane L (0..31) holds 16 K values (two runs of 8) of
// column N = nt*16 + (L&15), K = kt*32 + ((e>>3)<<4) + ((L>>4)<<3) + (e&7)   (ISA 7.12.2 B layout)
// ---------------------------------------------------------------------------
__global__ void pack_weights(const float* __restrict__ W, _Float16* __restrict__ P,
                             int ldn, int ntiles, int total) {
    int idx = blockIdx.x * blockDim.x + threadIdx.x;
    if (idx >= total) return;
    int tile = idx >> 9;          // 512 halves per tile
    int r    = idx & 511;
    int L = r >> 4, e = r & 15;
    int nt = tile % ntiles, kt = tile / ntiles;
    int N = nt * 16 + (L & 15);
    int K = kt * 32 + ((e >> 3) << 4) + ((L >> 4) << 3) + (e & 7);
    P[idx] = (_Float16)W[(size_t)K * ldn + N];
}

// Build an A-fragment (16x32 f16) from an LDS row-major f16 tile.
// Lane L: M = row (caller supplies), element e: K = kt*32 + ((e>>3)<<4) + ((L>>4)<<3) + (e&7)
// => two contiguous 8-half (16B) runs per lane.
__device__ __forceinline__ v16h a_frag(const _Float16* xs, int ldk, int M, int kt, int lane) {
    int hi = lane >> 4;
    int K0 = kt * 32 + hi * 8;
    const v8h* p0 = (const v8h*)(xs + M * ldk + K0);
    const v8h* p1 = (const v8h*)(xs + M * ldk + K0 + 16);
    return __builtin_shufflevector(*p0, *p1, 0,1,2,3,4,5,6,7,8,9,10,11,12,13,14,15);
}

// ---------------------------------------------------------------------------
// Fused shifted-window attention. 256 threads, 4 windows (32 rows) per block.
// Dynamic LDS 64KB: qkvh f16[32][768] (48KB) + xs f16[32][256] (16KB).
// Shift roll folded into gather (stage 1) and scatter (stage 4) indices.
// ---------------------------------------------------------------------------
__global__ __launch_bounds__(256) void swin_fused(
    const float* __restrict__ x,
    const float* __restrict__ qkv_b,
    const float* __restrict__ proj_b,
    const float* __restrict__ rel_bias,
    const _Float16* __restrict__ wq_packed,
    const _Float16* __restrict__ wp_packed,
    float* __restrict__ out) {

    extern __shared__ char smem[];
    _Float16* qkvh = (_Float16*)smem;                       // [32][768]
    _Float16* xs   = (_Float16*)(smem + ROWS * 768 * 2);    // [32][256], reused for attn out

    const int t    = threadIdx.x;
    const int lane = t & 31;
    const int wv   = t >> 5;
    const int gw0  = blockIdx.x * WPB;

    // ---- Stage 1: gather rolled x rows, convert f32 -> f16 into LDS --------
    {
        int row = t >> 3;                 // 0..31
        int col = (t & 7) * 32;           // 0..224
        int gw  = gw0 + (row >> 3);
        int bb  = gw >> 10;
        int tok = (((gw & 1023) * WS) + (row & 7) + SHIFTV) & (TSEQ - 1);
        const float4* src = (const float4*)(x + ((size_t)bb * TSEQ + tok) * DIMD + col);
        _Float16* dst = xs + row * DIMD + col;
        #pragma unroll
        for (int i = 0; i < 8; ++i) {
            float4 v = src[i];
            v4h h;
            h[0] = (_Float16)v.x; h[1] = (_Float16)v.y;
            h[2] = (_Float16)v.z; h[3] = (_Float16)v.w;
            *(v4h*)(dst + 4 * i) = h;
        }
    }
    __syncthreads();

    // ---- Stage 2: QKV GEMM [32,256] x [256,768] via WMMA -------------------
    {
        int mt = wv & 1;
        int M  = mt * 16 + (lane & 15);
        #pragma unroll
        for (int i = 0; i < 12; ++i) {
            int nt = (wv >> 1) + i * 4;   // wave-group covers 12 of 48 N-tiles
            if (i + 1 < 12)
                __builtin_prefetch(wq_packed + ((size_t)(nt + 4)) * 512 + lane * 16, 0, 0);
            v8f acc = {};
            #pragma unroll
            for (int kt = 0; kt < KTILES; ++kt) {
                v16h a = a_frag(xs, DIMD, M, kt, lane);
                v16h bfrag = *(const v16h*)(wq_packed + ((size_t)kt * NT_QKV + nt) * 512 + lane * 16);
                acc = __builtin_amdgcn_wmma_f32_16x16x32_f16(
                          false, a, false, bfrag, (short)0, acc, false, false);
            }
            float bias = qkv_b[nt * 16 + (lane & 15)];
            int N = nt * 16 + (lane & 15);
            #pragma unroll
            for (int r = 0; r < 8; ++r) {
                int Mo = mt * 16 + r + ((lane >> 4) << 3);   // C/D layout
                qkvh[Mo * 768 + N] = (_Float16)(acc[r] + bias);
            }
        }
    }
    __syncthreads();

    // ---- Stage 3: windowed softmax attention (f32 math in LDS) -------------
    {
        int win  = t >> 6;        // 0..3
        int head = (t >> 3) & 7;  // 0..7
        int qi   = t & 7;         // 0..7
        const float scale = 0.17677669529663688f;  // 32^-0.5
        const _Float16* qp = qkvh + (win * WS + qi) * 768 + head * HDIM;
        float s[WS];
        float mx = -1e30f;
        #pragma unroll
        for (int j = 0; j < WS; ++j) {
            const _Float16* kp = qkvh + (win * WS + j) * 768 + 256 + head * HDIM;
            float d = 0.f;
            #pragma unroll
            for (int dd = 0; dd < HDIM; ++dd) d += (float)qp[dd] * (float)kp[dd];
            d = d * scale + rel_bias[(j - qi + WS - 1) * HEADS + head];
            s[j] = d;
            mx = fmaxf(mx, d);
        }
        float sum = 0.f;
        #pragma unroll
        for (int j = 0; j < WS; ++j) { s[j] = __expf(s[j] - mx); sum += s[j]; }
        float inv = 1.f / sum;
        #pragma unroll
        for (int j = 0; j < WS; ++j) s[j] *= inv;
        _Float16* op = xs + (win * WS + qi) * DIMD + head * HDIM;  // reuse xs
        #pragma unroll
        for (int dd = 0; dd < HDIM; ++dd) {
            float a = 0.f;
            #pragma unroll
            for (int j = 0; j < WS; ++j)
                a += s[j] * (float)qkvh[(win * WS + j) * 768 + 512 + head * HDIM + dd];
            op[dd] = (_Float16)a;
        }
    }
    __syncthreads();

    // ---- Stage 4: Proj GEMM [32,256] x [256,256] + rolled scatter ----------
    {
        int mt = wv & 1;
        int M  = mt * 16 + (lane & 15);
        #pragma unroll
        for (int i = 0; i < 4; ++i) {
            int nt = (wv >> 1) + i * 4;   // 4 of 16 N-tiles per wave-group
            v8f acc = {};
            #pragma unroll
            for (int kt = 0; kt < KTILES; ++kt) {
                v16h a = a_frag(xs, DIMD, M, kt, lane);
                v16h bfrag = *(const v16h*)(wp_packed + ((size_t)kt * NT_PROJ + nt) * 512 + lane * 16);
                acc = __builtin_amdgcn_wmma_f32_16x16x32_f16(
                          false, a, false, bfrag, (short)0, acc, false, false);
            }
            float bias = proj_b[nt * 16 + (lane & 15)];
            int N = nt * 16 + (lane & 15);
            #pragma unroll
            for (int r = 0; r < 8; ++r) {
                int Ml  = mt * 16 + r + ((lane >> 4) << 3);
                int gw  = gw0 + (Ml >> 3);
                int bb  = gw >> 10;
                int tok = (((gw & 1023) * WS) + (Ml & 7) + SHIFTV) & (TSEQ - 1);
                out[((size_t)bb * TSEQ + tok) * DIMD + N] = acc[r] + bias;
            }
        }
    }
}

extern "C" void kernel_launch(void* const* d_in, const int* in_sizes, int n_in,
                              void* d_out, int out_size, void* d_ws, size_t ws_size,
                              hipStream_t stream) {
    const float* x        = (const float*)d_in[0];
    const float* qkv_w    = (const float*)d_in[1];
    const float* qkv_b    = (const float*)d_in[2];
    const float* proj_w   = (const float*)d_in[3];
    const float* proj_b   = (const float*)d_in[4];
    const float* rel_bias = (const float*)d_in[5];

    const int totq = KTILES * NT_QKV * 512;   // 196608 halves = 384KB
    const int totp = KTILES * NT_PROJ * 512;  //  65536 halves = 128KB
    _Float16* wq_packed = (_Float16*)d_ws;
    _Float16* wp_packed = (_Float16*)((char*)d_ws + (size_t)totq * sizeof(_Float16));

    pack_weights<<<(totq + 255) / 256, 256, 0, stream>>>(qkv_w, wq_packed, 768, NT_QKV, totq);
    pack_weights<<<(totp + 255) / 256, 256, 0, stream>>>(proj_w, wp_packed, 256, NT_PROJ, totp);

    swin_fused<<<NWIN_TOT / WPB, 256, 65536, stream>>>(
        x, qkv_b, proj_b, rel_bias, wq_packed, wp_packed, (float*)d_out);
}
